// ProductVectorQuantizer_15710990369637
// MI455X (gfx1250) — matrix operand order repaired and checked by reading, not compiled, not run on hardware
//
#include <hip/hip_runtime.h>
#include <hip/hip_bf16.h>
#include <math.h>
#include <stdint.h>

typedef __attribute__((ext_vector_type(16))) __bf16 v16bf;
typedef __attribute__((ext_vector_type(8)))  float  v8f;
typedef __attribute__((ext_vector_type(4)))  unsigned int u32x4;
typedef __attribute__((ext_vector_type(4)))  int i32x4;
typedef __attribute__((ext_vector_type(8)))  int i32x8;

#define N_TOK   65536
#define D_DIM   512
#define H_HEADS 8
#define K_CODES 512
#define HD_DIM  64
#define ROWS_PER_WG 128
#define MAIN_BLOCKS (N_TOK / ROWS_PER_WG)   // 512

// workspace layout (bytes)
#define WS_CNORM  0                // 4096 f32  (16 KB)
#define WS_COUNTS 16384            // 4096 u32  (16 KB)
#define WS_PLOSS  32768            // 512 f32   (2 KB)
#define WS_BBF16  34816            // 262144 bf16 (512 KB), B-layout swizzled

#if defined(__AMDGCN__) && __has_builtin(__builtin_amdgcn_tensor_load_to_lds) && \
    __has_builtin(__builtin_amdgcn_s_wait_tensorcnt)
#define USE_TDM 1
#else
#define USE_TDM 0
#endif

// ---------------------------------------------------------------------------
// Prep: codebook norms + bf16 codebooks pre-swizzled into WMMA B-matrix layout
// B (16-bit, 32x16): lane = hs*16 + col(=codeword%16), elem e -> dim = e + 16*hs (+32*chunk)
// ---------------------------------------------------------------------------
__global__ void pvq_prep(const float* __restrict__ cb, float* __restrict__ cnorm,
                         unsigned* __restrict__ counts, __bf16* __restrict__ bbf) {
  int id = blockIdx.x * blockDim.x + threadIdx.x;       // 0..4095 (h*K + cw)
  if (id >= H_HEADS * K_CODES) return;
  counts[id] = 0u;
  int h = id / K_CODES, cw = id % K_CODES;
  const float* row = cb + (size_t)id * HD_DIM;
  int cw_tile = cw >> 4, col = cw & 15;
  __bf16* bh = bbf + (size_t)h * (K_CODES * HD_DIM);
  float nrm = 0.f;
  for (int d = 0; d < HD_DIM; ++d) {
    float v = row[d];
    nrm += v * v;
    int chunk = d >> 5, rem = d & 31;
    int hs = rem >> 4, e = rem & 15;                    // dim = chunk*32 + hs*16 + e
    int lane = hs * 16 + col;
    bh[((((cw_tile * 2 + chunk) * 32) + lane) << 4) + e] = (__bf16)v;
  }
  cnorm[id] = nrm;
}

// ---------------------------------------------------------------------------
// Main: per-WG 128 rows; loop heads; bf16 WMMA distance scores + fused argmin
// ---------------------------------------------------------------------------
__global__ __launch_bounds__(256) void pvq_main(
    const float* __restrict__ z_e, const float* __restrict__ cb,
    const float* __restrict__ cnorm, const __bf16* __restrict__ bbf,
    unsigned* __restrict__ counts, float* __restrict__ ploss,
    float* __restrict__ out_zq, float* __restrict__ out_idx) {

  __shared__ __attribute__((aligned(32))) __bf16 ldsB[32 * 2 * 32 * 16]; // 64 KB
  __shared__ __attribute__((aligned(32))) __bf16 ldsA[8 * 2 * 32 * 16];  // 16 KB
  __shared__ float cnLds[K_CODES];                                        // 2 KB
  __shared__ int   idxLds[8][16];
  __shared__ float redLds[256];

  const int t = threadIdx.x;
  const int wave = t >> 5, lane = t & 31;
  const int col = lane & 15, hs = lane >> 4;
  const int blockRow0 = blockIdx.x * ROWS_PER_WG;
  float lossAcc = 0.f;

  for (int h = 0; h < H_HEADS; ++h) {
    __syncthreads();

    // ---- stage B head (64 KB, already in VGPR layout) ----
#if USE_TDM
    if (wave == 0) {
      // Tensor Data Mover: one 1-D 64KB tile, global -> LDS, 8-byte elements.
      unsigned long long ga = (unsigned long long)(uintptr_t)bbf
                            + (unsigned long long)h * (K_CODES * HD_DIM * 2);
      unsigned ldsb = (unsigned)(uintptr_t)&ldsB[0];
      u32x4 g0;
      g0[0] = 1u;                                        // count=1, user D#
      g0[1] = ldsb;                                      // lds_addr
      g0[2] = (unsigned)(ga & 0xffffffffull);            // global_addr[31:0]
      g0[3] = (unsigned)((ga >> 32) & 0x01ffffffull)     // global_addr[56:32]
            | (2u << 30);                                // type=2 ("image")
      i32x8 g1;
      g1[0] = 0x00030000;          // data_size=3 (8B), no flags
      g1[1] = (int)(8192u << 16);  // tensor_dim0[15:0]=8192 (8B units)
      g1[2] = (int)(1u << 16);     // tensor_dim0[31:16]=0 | tensor_dim1=1
      g1[3] = (int)(8192u << 16);  // tensor_dim1 hi=0 | tile_dim0=8192
      g1[4] = 0;                   // tile_dim1=0 (unused), tile_dim2=0
      g1[5] = 8192;                // tensor_dim0_stride[31:0]
      g1[6] = 0;                   // stride0 hi | tensor_dim1_stride lo
      g1[7] = 0;
      i32x4 gz4 = (i32x4){0, 0, 0, 0};
      i32x8 gz8 = (i32x8){0, 0, 0, 0, 0, 0, 0, 0};
      __builtin_amdgcn_tensor_load_to_lds(g0, g1, gz4, gz4, gz8, 0);
      __builtin_amdgcn_s_wait_tensorcnt(0);
    }
#else
    {
      const uint4* src = (const uint4*)(bbf + (size_t)h * (K_CODES * HD_DIM));
      uint4* dst = (uint4*)ldsB;
#pragma unroll
      for (int i = 0; i < 16; ++i) dst[t + i * 256] = src[t + i * 256];
    }
#endif

    cnLds[t]       = cnorm[h * K_CODES + t];
    cnLds[t + 256] = cnorm[h * K_CODES + t + 256];
    // stage A: 128 rows x 64 dims, f32 -> bf16, scatter into A-matrix layout
#pragma unroll
    for (int i = 0; i < 32; ++i) {
      int idx = i * 256 + t;
      int rl = idx >> 6, d = idx & 63;
      float v = z_e[(size_t)(blockRow0 + rl) * D_DIM + h * HD_DIM + d];
      int chunk = d >> 5, rem = d & 31;
      int ahs, ae;
      ahs = rem >> 4; ae = rem & 15;
      int alane = ahs * 16 + (rl & 15);
      ldsA[(((((rl >> 4) * 2 + chunk) * 32) + alane) << 4) + ae] = (__bf16)v;
    }
    __syncthreads();

    const v16bf* pA = (const v16bf*)ldsA;
    const v16bf* pB = (const v16bf*)ldsB;
    v16bf a0 = pA[(wave * 2 + 0) * 32 + lane];
    v16bf a1 = pA[(wave * 2 + 1) * 32 + lane];

    float bestS[8]; int bestI[8];
#pragma unroll
    for (int r = 0; r < 8; ++r) { bestS[r] = 3.4e38f; bestI[r] = 0; }

#pragma unroll 2
    for (int tile = 0; tile < 32; ++tile) {
      v16bf b0 = pB[(tile * 2 + 0) * 32 + lane];
      v16bf b1 = pB[(tile * 2 + 1) * 32 + lane];
      v8f acc = {};
      acc = __builtin_amdgcn_wmma_f32_16x16x32_bf16(false, a0, false, b0,
                                                    (short)0, acc, false, false);
      acc = __builtin_amdgcn_wmma_f32_16x16x32_bf16(false, a1, false, b1,
                                                    (short)0, acc, false, false);
      float cn = cnLds[tile * 16 + col];
      int cwi = tile * 16 + col;
#pragma unroll
      for (int r = 0; r < 8; ++r) {
        float s = cn - 2.f * acc[r];                    // ||c||^2 - 2 z.c
        if (s < bestS[r]) { bestS[r] = s; bestI[r] = cwi; }   // candidates ascend -> keeps first
      }
    }

    // branchless butterfly argmin: monotonic u64 key = sortable(score)<<32 | idx
    unsigned long long key[8];
#pragma unroll
    for (int r = 0; r < 8; ++r) {
      unsigned sb = __float_as_uint(bestS[r]);
      sb = (sb & 0x80000000u) ? ~sb : (sb | 0x80000000u);
      key[r] = ((unsigned long long)sb << 32) | (unsigned)bestI[r];
    }
#pragma unroll
    for (int bit = 1; bit < 16; bit <<= 1) {
#pragma unroll
      for (int r = 0; r < 8; ++r) {
        unsigned long long ok = __shfl_xor(key[r], bit, 32);
        key[r] = ok < key[r] ? ok : key[r];
      }
    }
    if (col == 0) {
#pragma unroll
      for (int r = 0; r < 8; ++r)
        idxLds[wave][hs * 8 + r] = (int)(unsigned)(key[r] & 0xffffffffull);
    }
    __syncthreads();

    // write phase: wave covers its 16 rows; 2 f32 dims per lane (coalesced)
    for (int rr = 0; rr < 16; ++rr) {
      int bidx = idxLds[wave][rr];
      size_t row = (size_t)blockRow0 + wave * 16 + rr;
      const float* crow = cb + ((size_t)h * K_CODES + bidx) * HD_DIM;
      int d0 = lane * 2;
      float q0 = crow[d0], q1 = crow[d0 + 1];
      size_t off = row * D_DIM + h * HD_DIM + d0;
      float e0 = z_e[off], e1 = z_e[off + 1];
      out_zq[off] = q0; out_zq[off + 1] = q1;
      lossAcc += (q0 - e0) * (q0 - e0) + (q1 - e1) * (q1 - e1);
      if (lane == 0) {
        out_idx[row * H_HEADS + h] = (float)bidx;
        atomicAdd(&counts[h * K_CODES + bidx], 1u);
      }
    }
  }

  // deterministic per-WG loss reduction
  __syncthreads();
  redLds[t] = lossAcc;
  __syncthreads();
  for (int s2 = 128; s2 > 0; s2 >>= 1) {
    if (t < s2) redLds[t] += redLds[t + s2];
    __syncthreads();
  }
  if (t == 0) ploss[blockIdx.x] = redLds[0];
}

// ---------------------------------------------------------------------------
// Finalize: losses + perplexity (single block, deterministic)
// ---------------------------------------------------------------------------
__global__ void pvq_final(const float* __restrict__ ploss,
                          const unsigned* __restrict__ counts,
                          float* __restrict__ scal) {
  __shared__ float red[256];
  int t = threadIdx.x;
  red[t] = ploss[t] + ploss[t + 256];
  __syncthreads();
  for (int k = 128; k > 0; k >>= 1) { if (t < k) red[t] += red[t + k]; __syncthreads(); }
  float closs = red[0] / (float)((size_t)N_TOK * D_DIM);

  float perpSum = 0.f;
  for (int h = 0; h < H_HEADS; ++h) {
    float part = 0.f;
    for (int i = t; i < K_CODES; i += 256) {
      float p = (float)counts[h * K_CODES + i] * (1.f / (float)N_TOK);
      part += p * logf(p + 1e-10f);
    }
    __syncthreads();
    red[t] = part;
    __syncthreads();
    for (int k = 128; k > 0; k >>= 1) { if (t < k) red[t] += red[t + k]; __syncthreads(); }
    if (t == 0) perpSum += expf(-red[0]);
    __syncthreads();
  }
  if (t == 0) {
    scal[0] = closs;            // codebook_loss
    scal[1] = 0.1f * closs;     // commitment_loss
    scal[2] = perpSum / (float)H_HEADS;
  }
}

extern "C" void kernel_launch(void* const* d_in, const int* in_sizes, int n_in,
                              void* d_out, int out_size, void* d_ws, size_t ws_size,
                              hipStream_t stream) {
  (void)in_sizes; (void)n_in; (void)out_size; (void)ws_size;
  const float* z_e = (const float*)d_in[0];
  const float* cb  = (const float*)d_in[1];
  char* ws = (char*)d_ws;
  float*    cnorm  = (float*)(ws + WS_CNORM);
  unsigned* counts = (unsigned*)(ws + WS_COUNTS);
  float*    pls    = (float*)(ws + WS_PLOSS);
  __bf16*   bbf    = (__bf16*)(ws + WS_BBF16);

  float* out     = (float*)d_out;
  float* out_zq  = out;
  float* out_idx = out + (size_t)N_TOK * D_DIM;
  float* scal    = out_idx + (size_t)N_TOK * H_HEADS;

  pvq_prep<<<16, 256, 0, stream>>>(cb, cnorm, counts, bbf);
  pvq_main<<<MAIN_BLOCKS, 256, 0, stream>>>(z_e, cb, cnorm, bbf, counts, pls,
                                            out_zq, out_idx);
  pvq_final<<<1, 256, 0, stream>>>(pls, counts, scal);
}
